// ProtodynModel_80762565034188
// MI455X (gfx1250) — matrix operand order repaired and checked by reading, not compiled, not run on hardware
//
#include <hip/hip_runtime.h>
#include <hip/hip_fp16.h>

typedef __attribute__((ext_vector_type(16))) _Float16 v16h;
typedef __attribute__((ext_vector_type(8)))  float    v8f;
typedef __attribute__((ext_vector_type(4)))  float    v4f;
typedef __attribute__((ext_vector_type(4)))  unsigned int v4u;
typedef __attribute__((ext_vector_type(8)))  int      v8i;
typedef __attribute__((ext_vector_type(4)))  int      v4i;

#define BN_INV 0.9999950000374997f   // 1/sqrt(1+1e-5)
#define WGSZ 256

__constant__ float CHI_TAB[20][4] = {
  {0,0,0,0},{1,1,1,1},{1,1,0,0},{1,1,0,0},{1,0,0,0},
  {1,1,1,0},{1,1,1,0},{0,0,0,0},{1,1,0,0},{1,1,0,0},
  {1,1,0,0},{1,1,1,1},{1,1,1,0},{1,1,0,0},{0,0,0,0},
  {1,0,0,0},{1,0,0,0},{1,1,0,0},{1,1,0,0},{1,0,0,0}};

__device__ __forceinline__ void atomicMaxFloat(float* addr, float val) {
  if (val >= 0.0f) atomicMax((int*)addr, __float_as_int(val));
  else             atomicMin((unsigned int*)addr, (unsigned int)__float_as_int(val));
}

// ---------------------------------------------------------------------------
// TDM: DMA a 64x64 f32 tile (row-major, stride 64) from global into LDS.
// D# fields per CDNA5 ISA (count=1, data_size=4B,
// tensor_dim0/1 = 64, tile_dim0/1 = 64, tensor_dim0_stride = 64, type=2).
// 6-arg builtin on this toolchain: (v4u, v8i, v4i, v4i, v8i, cpol).
// ---------------------------------------------------------------------------
__device__ __forceinline__ void tdm_load_w64x64(const float* W, unsigned int lds_off) {
  unsigned long long ga = (unsigned long long)(uintptr_t)W;
  v4u g0;
  g0[0] = 1u;                                                   // count=1, user mode
  g0[1] = lds_off;                                              // lds_addr
  g0[2] = (unsigned int)(ga & 0xFFFFFFFFull);                   // global_addr lo
  g0[3] = (unsigned int)((ga >> 32) & 0x01FFFFFFull) | (2u << 30); // addr hi | type=2
  v8i g1;
  g1[0] = (int)(2u << 16);    // wg_mask=0, data_size=2 (4 bytes)
  g1[1] = (int)(64u << 16);   // tensor_dim0=64 (bits 79:48, low half)
  g1[2] = (int)(64u << 16);   // tensor_dim0 hi=0 | tensor_dim1=64 (low half)
  g1[3] = (int)(64u << 16);   // tensor_dim1 hi=0 | tile_dim0=64
  g1[4] = (int)(64u);         // tile_dim1=64 | tile_dim2=0
  g1[5] = (int)(64u);         // tensor_dim0_stride=64 (low 32)
  g1[6] = 0;                  // stride hi | tensor_dim1_stride lo
  g1[7] = 0;
  v4i gz4 = {0, 0, 0, 0};
  v8i gz8 = {0, 0, 0, 0, 0, 0, 0, 0};
  __builtin_amdgcn_tensor_load_to_lds(g0, g1, gz4, gz4, gz8, 0);
}

// ---------------------------------------------------------------------------
// Y[n,64] = act( bn( X[n,64] @ W[64,64] + bias ) )   via v_wmma_f32_16x16x32_f16
// W staged once per block: TDM -> LDS f32, then transposed f16 [col][k] so each
// B fragment is one aligned 32B LDS read. One wave = 16-row M tile, 8 WMMAs.
// ---------------------------------------------------------------------------
template <bool HAS_BIAS, bool HAS_BN, bool RELU>
__global__ __launch_bounds__(WGSZ) void gemm64_wmma_kernel(
    const float* __restrict__ X, const float* __restrict__ W,
    const float* __restrict__ bias, const float* __restrict__ bn_g,
    const float* __restrict__ bn_beta, float* __restrict__ Y, int n) {
  __shared__ float wf32[64 * 64];      // 16KB raw f32 tile (TDM destination)
  __shared__ v16h  whv[64 * 4];        // 8KB transposed f16: [col][k], 64 halves/col

  // --- stage W: wave 0 issues the tensor DMA, whole block converts ---
  if ((threadIdx.x >> 5) == 0) {
    tdm_load_w64x64(W, (unsigned int)(uintptr_t)wf32);
    __builtin_amdgcn_s_wait_tensorcnt(0);
  }
  __syncthreads();
  {
    _Float16* wh = (_Float16*)whv;
    for (int i = threadIdx.x; i < 4096; i += WGSZ) {
      int k = i >> 6, col = i & 63;
      wh[col * 64 + k] = (_Float16)wf32[i];
    }
  }
  __syncthreads();

  const int lane = threadIdx.x & 31;
  const int wave = threadIdx.x >> 5;
  const int row0 = (blockIdx.x * (WGSZ >> 5) + wave) * 16;
  if (row0 < n) {                     // whole-wave predicate: EXEC all-ones inside
    const int mlane = lane & 15;
    const int hi    = lane >> 4;
    const int mrow  = row0 + mlane;
    const int ldrow = (mrow < n) ? mrow : (n - 1);
    const bool full = (row0 + 16 <= n);   // wave-uniform
    const v4f* Xv = (const v4f*)(X + (size_t)ldrow * 64);

    // A fragments (contiguous k = 16*hi + s, matching B), 4x b128 loads per K-tile
    v16h a[2];
#pragma unroll
    for (int kt = 0; kt < 2; ++kt) {
      v4f p0 = Xv[kt * 8 + 4 * hi + 0];
      v4f p1 = Xv[kt * 8 + 4 * hi + 1];
      v4f p2 = Xv[kt * 8 + 4 * hi + 2];
      v4f p3 = Xv[kt * 8 + 4 * hi + 3];
#pragma unroll
      for (int q = 0; q < 4; ++q) {
        a[kt][q]      = (_Float16)p0[q];
        a[kt][q + 4]  = (_Float16)p1[q];
        a[kt][q + 8]  = (_Float16)p2[q];
        a[kt][q + 12] = (_Float16)p3[q];
      }
    }

#pragma unroll
    for (int nt = 0; nt < 4; ++nt) {
      const int col = nt * 16 + mlane;
      v8f c = {};
#pragma unroll
      for (int kt = 0; kt < 2; ++kt) {
        v16h b = whv[col * 4 + kt * 2 + hi];   // 32B aligned LDS read
        c = __builtin_amdgcn_wmma_f32_16x16x32_f16(false, a[kt], false, b,
                                                   (short)0, c, false, false);
      }
      float addv = HAS_BIAS ? bias[col] : 0.0f;
      float gv = 1.0f, bv = 0.0f;
      if (HAS_BN) { gv = bn_g[col]; bv = bn_beta[col]; }
      // epilogue: single base pointer, constant store offsets on the fast path
      float* yp = Y + (size_t)(row0 + 8 * hi) * 64 + col;
      if (full) {
#pragma unroll
        for (int r = 0; r < 8; ++r) {
          float v = c[r] + addv;
          if (HAS_BN) v = v * BN_INV * gv + bv;
          if (RELU)   v = fmaxf(v, 0.0f);
          yp[(size_t)r * 64] = v;
        }
      } else {
#pragma unroll
        for (int r = 0; r < 8; ++r) {
          int row = row0 + r + 8 * hi;
          float v = c[r] + addv;
          if (HAS_BN) v = v * BN_INV * gv + bv;
          if (RELU)   v = fmaxf(v, 0.0f);
          if (row < n) yp[(size_t)r * 64] = v;
        }
      }
    }
  }
}

// ---------------------------------------------------------------------------
__global__ void embed_kernel(const float* __restrict__ X, const float* __restrict__ W,
                             const float* __restrict__ b, float* __restrict__ Y,
                             int n, int fin) {
  int t = blockIdx.x * blockDim.x + threadIdx.x;
  if (t >= n * 64) return;
  int i = t >> 6, j = t & 63;
  float acc = b[j];
  for (int k = 0; k < fin; ++k) acc = fmaf(X[(size_t)i * fin + k], W[(size_t)k * 64 + j], acc);
  Y[t] = fmaxf(acc, 0.0f);
}

__global__ void rel_gate_kernel(const float* __restrict__ attr, const int* __restrict__ edges,
                                const int* __restrict__ resid, const float* __restrict__ table,
                                const float* __restrict__ W1, const float* __restrict__ b1,
                                const float* __restrict__ W2, const float* __restrict__ b2,
                                float* __restrict__ ea, int ne) {
  int e = blockIdx.x * blockDim.x + threadIdx.x;
  if (e >= ne) return;
  int s = edges[e], d = edges[ne + e];
  const float* rs = table + resid[s] * 16;
  const float* rd = table + resid[d] * 16;
  float acc = b2[0];
#pragma unroll 4
  for (int j = 0; j < 16; ++j) {
    float h = b1[j];
    for (int k = 0; k < 16; ++k) h = fmaf(rs[k], W1[k * 16 + j], h);
    for (int k = 0; k < 16; ++k) h = fmaf(rd[k], W1[(16 + k) * 16 + j], h);
    acc = fmaf(fmaxf(h, 0.0f), W2[j], acc);
  }
  float gate = 1.0f / (1.0f + __expf(-acc));
#pragma unroll
  for (int k = 0; k < 8; ++k) ea[(size_t)e * 8 + k] = attr[(size_t)e * 8 + k] * gate;
}

__global__ void edge_update_kernel(float* __restrict__ ea,
                                   const float* __restrict__ W1, const float* __restrict__ b1,
                                   const float* __restrict__ g,  const float* __restrict__ beta,
                                   const float* __restrict__ W2, const float* __restrict__ b2,
                                   int ne) {
  int e = blockIdx.x * blockDim.x + threadIdx.x;
  if (e >= ne) return;
  float x[8], h[16];
#pragma unroll
  for (int k = 0; k < 8; ++k) x[k] = ea[(size_t)e * 8 + k];
#pragma unroll
  for (int j = 0; j < 16; ++j) {
    float a = b1[j];
#pragma unroll
    for (int k = 0; k < 8; ++k) a = fmaf(x[k], W1[k * 16 + j], a);
    a = a * BN_INV * g[j] + beta[j];
    h[j] = fmaxf(a, 0.0f);
  }
#pragma unroll
  for (int k = 0; k < 8; ++k) {
    float y = b2[k];
#pragma unroll
    for (int j = 0; j < 16; ++j) y = fmaf(h[j], W2[j * 8 + k], y);
    ea[(size_t)e * 8 + k] = y + x[k];
  }
}

__global__ void init_seg_kernel(float* __restrict__ emax, float* __restrict__ den, int n) {
  int i = blockIdx.x * blockDim.x + threadIdx.x;
  if (i >= n) return;
  emax[i] = __int_as_float(0xff800000);
  den[i]  = 0.0f;
}

__global__ void edge_score_kernel(const float* __restrict__ xl, const float* __restrict__ xr,
                                  const float* __restrict__ ea, const float* __restrict__ We,
                                  const float* __restrict__ att, const int* __restrict__ edges,
                                  int ne, float* __restrict__ esc, float* __restrict__ emax) {
  int e = blockIdx.x * blockDim.x + threadIdx.x;
  if (e >= ne) return;
  int s = edges[e], d = edges[ne + e];
  const float* pl = xl + (size_t)s * 64;
  const float* pr = xr + (size_t)d * 64;
  float ef[8];
  if (ea) {
#pragma unroll
    for (int k = 0; k < 8; ++k) ef[k] = ea[(size_t)e * 8 + k];
  }
  float sum = 0.0f;
#pragma unroll 8
  for (int j = 0; j < 64; ++j) {
    float m = pl[j] + pr[j];
    if (ea) {
#pragma unroll
      for (int k = 0; k < 8; ++k) m = fmaf(ef[k], We[k * 64 + j], m);
    }
    m = (m > 0.0f) ? m : 0.2f * m;
    sum = fmaf(m, att[j], sum);
  }
  esc[e] = sum;
  atomicMaxFloat(emax + d, sum);
}

__global__ void edge_exp_kernel(float* __restrict__ esc, const float* __restrict__ emax,
                                float* __restrict__ den, const int* __restrict__ edges, int ne) {
  int e = blockIdx.x * blockDim.x + threadIdx.x;
  if (e >= ne) return;
  int d = edges[ne + e];
  float ex = __expf(esc[e] - emax[d]);
  esc[e] = ex;
  atomicAdd(den + d, ex);
}

__global__ void edge_scatter_kernel(const float* __restrict__ xl, const float* __restrict__ esc,
                                    const float* __restrict__ den, const int* __restrict__ edges,
                                    int ne, float* __restrict__ out) {
  int t = blockIdx.x * blockDim.x + threadIdx.x;
  if (t >= ne * 64) return;
  int e = t >> 6, j = t & 63;
  int s = edges[e], d = edges[ne + e];
  float a = esc[e] / (den[d] + 1e-16f);
  atomicAdd(out + (size_t)d * 64 + j, a * xl[(size_t)s * 64 + j]);
}

__global__ void fusion_kernel(const float* __restrict__ sc, const float* __restrict__ sc_up,
                              const float* __restrict__ afW, const float* __restrict__ afb,
                              const float* __restrict__ afg, const float* __restrict__ afbeta,
                              float* __restrict__ bb, int n) {
  int i = blockIdx.x * blockDim.x + threadIdx.x;
  if (i >= n) return;
  float a = afb[0];
#pragma unroll 8
  for (int k = 0; k < 64; ++k) a = fmaf(sc[(size_t)i * 64 + k], afW[k], a);
  a = a * BN_INV * afg[0] + afbeta[0];
  a = 1.0f / (1.0f + __expf(-a));
#pragma unroll 8
  for (int j = 0; j < 64; ++j) bb[(size_t)i * 64 + j] += a * sc_up[(size_t)i * 64 + j];
}

__global__ void chi_mask_kernel(float* __restrict__ H, const int* __restrict__ resid,
                                int chi_idx, int n) {
  int t = blockIdx.x * blockDim.x + threadIdx.x;
  if (t >= n * 64) return;
  H[t] *= CHI_TAB[resid[t >> 6]][chi_idx];
}

__global__ void head_matvec_kernel(const float* __restrict__ X, const float* __restrict__ W,
                                   const float* __restrict__ b, float* __restrict__ Y,
                                   int n, int ko) {
  int t = blockIdx.x * blockDim.x + threadIdx.x;
  if (t >= n * ko) return;
  int i = t / ko, j = t % ko;
  float acc = b ? b[j] : 0.0f;
#pragma unroll 8
  for (int k = 0; k < 64; ++k) acc = fmaf(X[(size_t)i * 64 + k], W[k * ko + j], acc);
  Y[t] = acc;
}

// ---------------------------------------------------------------------------
static inline int cdiv(int a, int b) { return (a + b - 1) / b; }

static void launch_gemm(const float* X, const float* W, const float* bias,
                        const float* g, const float* beta, float* Y, int n,
                        bool relu, int blocks, hipStream_t stream) {
  if (g)               gemm64_wmma_kernel<true,  true,  true ><<<blocks, WGSZ, 0, stream>>>(X, W, bias, g, beta, Y, n);
  else if (bias&&relu) gemm64_wmma_kernel<true,  false, true ><<<blocks, WGSZ, 0, stream>>>(X, W, bias, g, beta, Y, n);
  else if (bias)       gemm64_wmma_kernel<true,  false, false><<<blocks, WGSZ, 0, stream>>>(X, W, bias, g, beta, Y, n);
  else if (relu)       gemm64_wmma_kernel<false, false, true ><<<blocks, WGSZ, 0, stream>>>(X, W, bias, g, beta, Y, n);
  else                 gemm64_wmma_kernel<false, false, false><<<blocks, WGSZ, 0, stream>>>(X, W, bias, g, beta, Y, n);
}

extern "C" void kernel_launch(void* const* d_in, const int* in_sizes, int n_in,
                              void* d_out, int out_size, void* d_ws, size_t ws_size,
                              hipStream_t stream) {
  const float* x_sc  = (const float*)d_in[0];
  const float* x_bb  = (const float*)d_in[1];
  const float* eattr = (const float*)d_in[2];
  const int* sc_e    = (const int*)d_in[3];
  const int* bb_e    = (const int*)d_in[4];
  const int* resid   = (const int*)d_in[5];
  const float* scW  = (const float*)d_in[6];  const float* scb  = (const float*)d_in[7];
  const float* bbW  = (const float*)d_in[8];  const float* bbb  = (const float*)d_in[9];
  const float* rtab = (const float*)d_in[10];
  const float* rrW1 = (const float*)d_in[11]; const float* rrb1 = (const float*)d_in[12];
  const float* rrW2 = (const float*)d_in[13]; const float* rrb2 = (const float*)d_in[14];
  const float* scWl = (const float*)d_in[15]; const float* scWr = (const float*)d_in[16];
  const float* scWe = (const float*)d_in[17]; const float* scAt = (const float*)d_in[18];
  const float* scRs = (const float*)d_in[19]; const float* scBi = (const float*)d_in[20];
  const float* bbWl = (const float*)d_in[21]; const float* bbWr = (const float*)d_in[22];
  const float* bbAt = (const float*)d_in[23]; const float* bbRs = (const float*)d_in[24];
  const float* bbBi = (const float*)d_in[25];
  const float* euW1 = (const float*)d_in[26]; const float* eub1 = (const float*)d_in[27];
  const float* eug  = (const float*)d_in[28]; const float* eube = (const float*)d_in[29];
  const float* euW2 = (const float*)d_in[30]; const float* eub2 = (const float*)d_in[31];
  const float* bnuW = (const float*)d_in[32]; const float* bnub = (const float*)d_in[33];
  const float* bnug = (const float*)d_in[34]; const float* bnube= (const float*)d_in[35];
  const float* afW  = (const float*)d_in[36]; const float* afb  = (const float*)d_in[37];
  const float* afg  = (const float*)d_in[38]; const float* afbe = (const float*)d_in[39];
  const float* chiW = (const float*)d_in[40]; const float* chib = (const float*)d_in[41];
  const float* adW1 = (const float*)d_in[42]; const float* adW2 = (const float*)d_in[43];
  const float* vcW1 = (const float*)d_in[44]; const float* vcb1 = (const float*)d_in[45];
  const float* vcW2 = (const float*)d_in[46]; const float* vcb2 = (const float*)d_in[47];
  const float* bhW1 = (const float*)d_in[48]; const float* bhb1 = (const float*)d_in[49];
  const float* phiW = (const float*)d_in[50]; const float* psiW = (const float*)d_in[51];
  const float* xcaW = (const float*)d_in[52]; const float* xcab = (const float*)d_in[53];
  const float* vcbW = (const float*)d_in[54]; const float* vcbb = (const float*)d_in[55];

  const int n  = in_sizes[5];
  const int ne = in_sizes[2] / 8;
  const int nb = in_sizes[4] / 2;
  const int L  = 12;

  float* ws = (float*)d_ws;
  size_t o = 0;
  float* scbuf[2]; scbuf[0] = ws + o; o += (size_t)n * 64; scbuf[1] = ws + o; o += (size_t)n * 64;
  float* bbbuf[2]; bbbuf[0] = ws + o; o += (size_t)n * 64; bbbuf[1] = ws + o; o += (size_t)n * 64;
  float* xl = ws + o; o += (size_t)n * 64;
  float* xr = ws + o; o += (size_t)n * 64;
  float* ea = ws + o; o += (size_t)ne * 8;
  float* esc = ws + o; o += (size_t)((ne > nb) ? ne : nb);
  float* emax = ws + o; o += n;
  float* den  = ws + o; o += n;

  const int gN64  = cdiv(n * 64, WGSZ);
  const int gN    = cdiv(n, WGSZ);
  const int gemmB = cdiv(cdiv(n, 16), 8);

  embed_kernel<<<gN64, WGSZ, 0, stream>>>(x_sc, scW, scb, scbuf[0], n, 21);
  embed_kernel<<<gN64, WGSZ, 0, stream>>>(x_bb, bbW, bbb, bbbuf[0], n, 12);
  rel_gate_kernel<<<cdiv(ne, WGSZ), WGSZ, 0, stream>>>(eattr, sc_e, resid, rtab,
                                                       rrW1, rrb1, rrW2, rrb2, ea, ne);

  int si = 0, bi = 0;
  for (int l = 0; l < L; ++l) {
    edge_update_kernel<<<cdiv(ne, WGSZ), WGSZ, 0, stream>>>(
        ea, euW1 + l * 128, eub1 + l * 16, eug + l * 16, eube + l * 16,
        euW2 + l * 128, eub2 + l * 8, ne);

    { // sidechain GAT
      const float* xin = scbuf[si]; float* xout = scbuf[si ^ 1];
      launch_gemm(xin, scWl + (size_t)l * 4096, nullptr, nullptr, nullptr, xl, n, false, gemmB, stream);
      launch_gemm(xin, scWr + (size_t)l * 4096, nullptr, nullptr, nullptr, xr, n, false, gemmB, stream);
      launch_gemm(xin, scRs + (size_t)l * 4096, scBi + l * 64, nullptr, nullptr, xout, n, false, gemmB, stream);
      init_seg_kernel<<<gN, WGSZ, 0, stream>>>(emax, den, n);
      edge_score_kernel<<<cdiv(ne, WGSZ), WGSZ, 0, stream>>>(xl, xr, ea, scWe + (size_t)l * 512,
                                                             scAt + l * 64, sc_e, ne, esc, emax);
      edge_exp_kernel<<<cdiv(ne, WGSZ), WGSZ, 0, stream>>>(esc, emax, den, sc_e, ne);
      edge_scatter_kernel<<<cdiv(ne * 64, WGSZ), WGSZ, 0, stream>>>(xl, esc, den, sc_e, ne, xout);
      si ^= 1;
    }
    { // backbone GAT
      const float* xin = bbbuf[bi]; float* xout = bbbuf[bi ^ 1];
      launch_gemm(xin, bbWl + (size_t)l * 4096, nullptr, nullptr, nullptr, xl, n, false, gemmB, stream);
      launch_gemm(xin, bbWr + (size_t)l * 4096, nullptr, nullptr, nullptr, xr, n, false, gemmB, stream);
      launch_gemm(xin, bbRs + (size_t)l * 4096, bbBi + l * 64, nullptr, nullptr, xout, n, false, gemmB, stream);
      init_seg_kernel<<<gN, WGSZ, 0, stream>>>(emax, den, n);
      edge_score_kernel<<<cdiv(nb, WGSZ), WGSZ, 0, stream>>>(xl, xr, nullptr, nullptr,
                                                             bbAt + l * 64, bb_e, nb, esc, emax);
      edge_exp_kernel<<<cdiv(nb, WGSZ), WGSZ, 0, stream>>>(esc, emax, den, bb_e, nb);
      edge_scatter_kernel<<<cdiv(nb * 64, WGSZ), WGSZ, 0, stream>>>(xl, esc, den, bb_e, nb, xout);
      bi ^= 1;
    }
  }

  float* sc = scbuf[si];
  float* bb = bbbuf[bi];
  float* out = (float*)d_out;

  launch_gemm(sc, bnuW, bnub, bnug, bnube, xl, n, true, gemmB, stream);
  fusion_kernel<<<gN, WGSZ, 0, stream>>>(sc, xl, afW, afb, afg, afbe, bb, n);

  for (int i = 0; i < 4; ++i) {
    launch_gemm(sc, chiW + (size_t)i * 4096, chib + i * 64, nullptr, nullptr, xr, n, true, gemmB, stream);
    chi_mask_kernel<<<gN64, WGSZ, 0, stream>>>(xr, resid, i, n);
    launch_gemm(xr, adW1, nullptr, nullptr, nullptr, xl, n, true, gemmB, stream);
    head_matvec_kernel<<<cdiv(n, WGSZ), WGSZ, 0, stream>>>(xl, adW2, nullptr, out + (size_t)i * n, n, 1);
  }
  launch_gemm(sc, vcW1, vcb1, nullptr, nullptr, xl, n, true, gemmB, stream);
  head_matvec_kernel<<<cdiv(n * 3, WGSZ), WGSZ, 0, stream>>>(xl, vcW2, vcb2, out + (size_t)4 * n, n, 3);
  launch_gemm(bb, bhW1 + 0 * 4096, bhb1 + 0 * 64, nullptr, nullptr, xl, n, true, gemmB, stream);
  head_matvec_kernel<<<cdiv(n, WGSZ), WGSZ, 0, stream>>>(xl, phiW, nullptr, out + (size_t)7 * n, n, 1);
  launch_gemm(bb, bhW1 + 1 * 4096, bhb1 + 1 * 64, nullptr, nullptr, xl, n, true, gemmB, stream);
  head_matvec_kernel<<<cdiv(n, WGSZ), WGSZ, 0, stream>>>(xl, psiW, nullptr, out + (size_t)8 * n, n, 1);
  launch_gemm(bb, bhW1 + 2 * 4096, bhb1 + 2 * 64, nullptr, nullptr, xl, n, true, gemmB, stream);
  head_matvec_kernel<<<cdiv(n * 3, WGSZ), WGSZ, 0, stream>>>(xl, xcaW, xcab, out + (size_t)9 * n, n, 3);
  launch_gemm(bb, bhW1 + 3 * 4096, bhb1 + 3 * 64, nullptr, nullptr, xl, n, true, gemmB, stream);
  head_matvec_kernel<<<cdiv(n * 3, WGSZ), WGSZ, 0, stream>>>(xl, vcbW, vcbb, out + (size_t)12 * n, n, 3);
}
